// Model_39238821216633
// MI455X (gfx1250) — 8-bit matrix operand order repaired and checked by reading, compiled, not run on hardware
//
#include <hip/hip_runtime.h>

typedef __attribute__((ext_vector_type(8))) int v8i;

union F8 {
    v8i  v;
    int  i[8];
    int2 d2[4];
    int4 d4[2];
};

#define BATCH 8192

__device__ __forceinline__ unsigned lds_addr_of(const void* p) {
    return (unsigned)(unsigned long long)(const __attribute__((address_space(3))) void*)p;
}

__device__ __forceinline__ void async_g2l_b128(unsigned lds_off, const void* gsrc) {
    asm volatile("global_load_async_to_lds_b128 %0, %1, off"
                 :: "v"(lds_off), "v"(gsrc)
                 : "memory");
}

__device__ __forceinline__ void wait_async0() {
    asm volatile("s_wait_asynccnt 0x0" ::: "memory");
}

__global__ __launch_bounds__(256) void prep_kernel(
    const float* __restrict__ w1, const float* __restrict__ w2,
    const float* __restrict__ w3, const float* __restrict__ wfc,
    signed char* __restrict__ w1s, signed char* __restrict__ w2s,
    signed char* __restrict__ w3s, signed char* __restrict__ wfct)
{
    int t = threadIdx.x;
    for (int i = t; i < 16 * 9; i += 256)
        w1s[i] = (signed char)(w1[i] >= 0.0f ? 1 : -1);
    for (int i = t; i < 32 * 192; i += 256) {
        int m = i / 192, k = i % 192, tap = k / 16, ic = k % 16;
        w2s[i] = (tap < 9) ? (signed char)(w2[m * 144 + ic * 9 + tap] >= 0.0f ? 1 : -1)
                           : (signed char)0;
    }
    for (int i = t; i < 32 * 320; i += 256) {
        int m = i / 320, k = i % 320, tap = k / 32, ic = k % 32;
        w3s[i] = (tap < 9) ? (signed char)(w3[m * 288 + ic * 9 + tap] >= 0.0f ? 1 : -1)
                           : (signed char)0;
    }
    for (int i = t; i < 3872 * 12; i += 256) {
        int f = i / 12, cls = i % 12;
        wfct[i] = (cls < 10) ? (signed char)(wfc[cls * 3872 + f] >= 0.0f ? 1 : -1)
                             : (signed char)0;
    }
}

__global__ __launch_bounds__(256) void conv1_kernel(
    const float* __restrict__ x, const signed char* __restrict__ w1s,
    signed char* __restrict__ h1)
{
    __shared__ float w1f[144];
    for (int i = threadIdx.x; i < 144; i += 256) w1f[i] = (float)w1s[i];
    __syncthreads();

    int idx = blockIdx.x * 256 + threadIdx.x;
    if (idx >= BATCH * 26 * 26) return;
    int b = idx / 676, r = idx % 676;
    int y = r / 26, xo = r % 26;

    const float* xp = x + (size_t)b * 784 + y * 28 + xo;
    float xv[9];
#pragma unroll
    for (int ky = 0; ky < 3; ++ky)
#pragma unroll
        for (int kx = 0; kx < 3; ++kx) xv[ky * 3 + kx] = xp[ky * 28 + kx];

    signed char* op = h1 + (size_t)idx * 16;
#pragma unroll
    for (int oc = 0; oc < 16; ++oc) {
        float acc = 0.0f;
#pragma unroll
        for (int tt = 0; tt < 9; ++tt) acc += w1f[oc * 9 + tt] * xv[tt];
        op[oc] = (signed char)(acc >= 0.0f ? 1 : -1);
    }
}

__global__ __launch_bounds__(32) void conv2_kernel(
    const signed char* __restrict__ h1, const signed char* __restrict__ w2s,
    signed char* __restrict__ h2)
{
    __shared__ signed char Bp[16 * 192];
    int lane = threadIdx.x;
    int tile = blockIdx.x;
    int b = tile / 36;
    int tp = (tile % 36) * 16;

    unsigned bbase = lds_addr_of(Bp);
    for (int i = lane; i < 144; i += 32) {
        int n = i / 9, t = i % 9;
        int p = tp + n;
        int py = p / 24, px = p % 24;
        int ky = t / 3, kx = t % 3;
        const signed char* src = h1 +
            (((size_t)b * 26 + py + ky) * 26 + (px + kx)) * 16;
        async_g2l_b128(bbase + (unsigned)(n * 192 + t * 16), src);
    }
    wait_async0();
    __syncthreads();

    int mrow = lane & 15;
    int alo  = (lane < 16) ? 0 : 8;
    int bko  = (lane < 16) ? 0 : 8;

    F8 c0 = {}; F8 c1 = {};
#pragma unroll
    for (int s = 0; s < 3; ++s) {
        int kb = s * 64;
        F8 a0, a1, bb;
        const signed char* ap0 = w2s + mrow * 192 + kb + alo;
        const signed char* ap1 = w2s + (16 + mrow) * 192 + kb + alo;
        a0.d2[0] = *(const int2*)(ap0);      a0.d2[1] = *(const int2*)(ap0 + 16);
        a0.d2[2] = *(const int2*)(ap0 + 32); a0.d2[3] = *(const int2*)(ap0 + 48);
        a1.d2[0] = *(const int2*)(ap1);      a1.d2[1] = *(const int2*)(ap1 + 16);
        a1.d2[2] = *(const int2*)(ap1 + 32); a1.d2[3] = *(const int2*)(ap1 + 48);
        const signed char* bp = Bp + mrow * 192 + kb + bko;
        bb.d2[0] = *(const int2*)(bp);      bb.d2[1] = *(const int2*)(bp + 16);
        bb.d2[2] = *(const int2*)(bp + 32); bb.d2[3] = *(const int2*)(bp + 48);
        c0.v = __builtin_amdgcn_wmma_i32_16x16x64_iu8(true, a0.v, true, bb.v, c0.v, false, false);
        c1.v = __builtin_amdgcn_wmma_i32_16x16x64_iu8(true, a1.v, true, bb.v, c1.v, false, false);
    }

    int p  = tp + (lane & 15);
    int py = p / 24, px = p % 24;
    signed char* op = h2 + (((size_t)b * 24 + py) * 24 + px) * 32;
    int mh = (lane < 16) ? 0 : 8;
#pragma unroll
    for (int v = 0; v < 8; ++v) {
        op[mh + v]      = (signed char)(c0.i[v] >= 0 ? 1 : -1);
        op[16 + mh + v] = (signed char)(c1.i[v] >= 0 ? 1 : -1);
    }
}

__global__ __launch_bounds__(32) void conv3_kernel(
    const signed char* __restrict__ h2, const signed char* __restrict__ w3s,
    signed char* __restrict__ h3)
{
    __shared__ signed char Bp[16 * 320];
    int lane = threadIdx.x;
    int tile = blockIdx.x;
    int b = tile / 31;
    int tp = (tile % 31) * 16;

    unsigned bbase = lds_addr_of(Bp);
    for (int i = lane; i < 144; i += 32) {
        int n = i / 9, t = i % 9;
        int p = tp + n; if (p > 483) p = 483;
        int py = p / 22, px = p % 22;
        int ky = t / 3, kx = t % 3;
        const signed char* src = h2 +
            (((size_t)b * 24 + py + ky) * 24 + (px + kx)) * 32;
        unsigned dst = bbase + (unsigned)(n * 320 + t * 32);
        async_g2l_b128(dst, src);
        async_g2l_b128(dst + 16, src + 16);
    }
    wait_async0();
    __syncthreads();

    int mrow = lane & 15;
    int alo  = (lane < 16) ? 0 : 8;
    int bko  = (lane < 16) ? 0 : 8;

    F8 c0 = {}; F8 c1 = {};
#pragma unroll
    for (int s = 0; s < 5; ++s) {
        int kb = s * 64;
        F8 a0, a1, bb;
        const signed char* ap0 = w3s + mrow * 320 + kb + alo;
        const signed char* ap1 = w3s + (16 + mrow) * 320 + kb + alo;
        a0.d2[0] = *(const int2*)(ap0);      a0.d2[1] = *(const int2*)(ap0 + 16);
        a0.d2[2] = *(const int2*)(ap0 + 32); a0.d2[3] = *(const int2*)(ap0 + 48);
        a1.d2[0] = *(const int2*)(ap1);      a1.d2[1] = *(const int2*)(ap1 + 16);
        a1.d2[2] = *(const int2*)(ap1 + 32); a1.d2[3] = *(const int2*)(ap1 + 48);
        const signed char* bp = Bp + mrow * 320 + kb + bko;
        bb.d2[0] = *(const int2*)(bp);      bb.d2[1] = *(const int2*)(bp + 16);
        bb.d2[2] = *(const int2*)(bp + 32); bb.d2[3] = *(const int2*)(bp + 48);
        c0.v = __builtin_amdgcn_wmma_i32_16x16x64_iu8(true, a0.v, true, bb.v, c0.v, false, false);
        c1.v = __builtin_amdgcn_wmma_i32_16x16x64_iu8(true, a1.v, true, bb.v, c1.v, false, false);
    }

    int p = tp + (lane & 15);
    if (p < 484) {
        int py = p / 22, px = p % 22;
        signed char* op = h3 + (((size_t)b * 22 + py) * 22 + px) * 32;
        int mh = (lane < 16) ? 0 : 8;
#pragma unroll
        for (int v = 0; v < 8; ++v) {
            op[mh + v]      = (signed char)(c0.i[v] >= 0 ? 1 : -1);
            op[16 + mh + v] = (signed char)(c1.i[v] >= 0 ? 1 : -1);
        }
    }
}

#define WB(w, j) ((int)(signed char)((w) >> (8 * (j))))

__global__ __launch_bounds__(256) void tail_kernel(
    const signed char* __restrict__ h3, const signed char* __restrict__ wfct,
    const float* __restrict__ bfc, float* __restrict__ out)
{
    __shared__ signed char W[3872 * 12];
    __shared__ float bf[10];
    for (int i = threadIdx.x; i < (3872 * 12) / 4; i += 256)
        ((int*)W)[i] = ((const int*)wfct)[i];
    if (threadIdx.x < 10) bf[threadIdx.x] = bfc[threadIdx.x];
    __syncthreads();

    int b = blockIdx.x * 256 + threadIdx.x;
    if (b >= BATCH) return;

    int acc[10];
#pragma unroll
    for (int c = 0; c < 10; ++c) acc[c] = 0;

    const signed char* hb = h3 + (size_t)b * 22 * 22 * 32;
    for (int c = 0; c < 32; ++c)
        for (int py = 0; py < 11; ++py)
            for (int px = 0; px < 11; ++px) {
                int y0 = 2 * py, x0 = 2 * px;
                int i00 = (y0 * 22 + x0) * 32 + c;
                int s4 = (int)hb[i00] + (int)hb[i00 + 32] +
                         (int)hb[i00 + 22 * 32] + (int)hb[i00 + 22 * 32 + 32];
                const signed char* wp = W + (c * 121 + py * 11 + px) * 12;
                int w0 = *(const int*)(wp);
                int w1 = *(const int*)(wp + 4);
                int w2 = *(const int*)(wp + 8);
                acc[0] += s4 * WB(w0, 0); acc[1] += s4 * WB(w0, 1);
                acc[2] += s4 * WB(w0, 2); acc[3] += s4 * WB(w0, 3);
                acc[4] += s4 * WB(w1, 0); acc[5] += s4 * WB(w1, 1);
                acc[6] += s4 * WB(w1, 2); acc[7] += s4 * WB(w1, 3);
                acc[8] += s4 * WB(w2, 0); acc[9] += s4 * WB(w2, 1);
            }

    float lg[10];
    float mx = -1e30f;
#pragma unroll
    for (int c = 0; c < 10; ++c) {
        lg[c] = 0.25f * (float)acc[c] + bf[c];
        mx = fmaxf(mx, lg[c]);
    }
    float s = 0.0f;
#pragma unroll
    for (int c = 0; c < 10; ++c) s += expf(lg[c] - mx);
    float lse = logf(s);
    float* op = out + (size_t)b * 10;
#pragma unroll
    for (int c = 0; c < 10; ++c) op[c] = lg[c] - mx - lse;
}

extern "C" void kernel_launch(void* const* d_in, const int* in_sizes, int n_in,
                              void* d_out, int out_size, void* d_ws, size_t ws_size,
                              hipStream_t stream)
{
    (void)in_sizes; (void)n_in; (void)out_size; (void)ws_size;
    const float* x   = (const float*)d_in[0];
    const float* w1  = (const float*)d_in[1];
    const float* w2  = (const float*)d_in[2];
    const float* w3  = (const float*)d_in[3];
    const float* wfc = (const float*)d_in[4];
    const float* bfc = (const float*)d_in[5];
    float* out = (float*)d_out;

    char* ws = (char*)d_ws;
    signed char* w1s  = (signed char*)(ws + 0);
    signed char* w2s  = (signed char*)(ws + 256);
    signed char* w3s  = (signed char*)(ws + 6400);
    signed char* wfct = (signed char*)(ws + 16640);
    signed char* h1   = (signed char*)(ws + 63232);
    signed char* h2   = h1 + (size_t)BATCH * 676 * 16;
    signed char* h3   = h2 + (size_t)BATCH * 576 * 32;

    prep_kernel<<<1, 256, 0, stream>>>(w1, w2, w3, wfc, w1s, w2s, w3s, wfct);
    conv1_kernel<<<(BATCH * 676 + 255) / 256, 256, 0, stream>>>(x, w1s, h1);
    conv2_kernel<<<BATCH * 36, 32, 0, stream>>>(h1, w2s, h2);
    conv3_kernel<<<BATCH * 31, 32, 0, stream>>>(h2, w3s, h3);
    tail_kernel<<<(BATCH + 255) / 256, 256, 0, stream>>>(h3, wfct, bfc, out);
}
